// AttentionV7_83829171683764
// MI455X (gfx1250) — matrix operand order repaired and checked by reading, not compiled, not run on hardware
//
#include <hip/hip_runtime.h>
#include <math.h>
#include <stdint.h>

// ---------------------------------------------------------------------------
// Types for CDNA5 WMMA (wave32): v_wmma_f32_16x16x32_bf16
// ---------------------------------------------------------------------------
typedef __attribute__((ext_vector_type(16))) __bf16 bf16x16;
typedef __attribute__((ext_vector_type(8)))  __bf16 bf16x8;
typedef __attribute__((ext_vector_type(8)))  float  f32x8;

static __device__ __forceinline__ f32x8 f32x8_zero() {
    f32x8 z;
#pragma unroll
    for (int i = 0; i < 8; ++i) z[i] = 0.0f;
    return z;
}

static __device__ __forceinline__ f32x8 wmma_bf16(bf16x16 a, bf16x16 b, f32x8 c) {
    // D = A(16x32 bf16) * B(32x16 bf16) + C(16x16 f32)
    return __builtin_amdgcn_wmma_f32_16x16x32_bf16(
        /*neg_a=*/false, a, /*neg_b=*/false, b,
        /*c_mod=*/(short)0, c, /*reuse_a=*/false, /*reuse_b=*/false);
}

// A-fragment (16x32): lane holds row M=lane%16; K runs {h*8..h*8+7, h*8+16..h*8+23}, h=lane/16.
static __device__ __forceinline__ bf16x16 load_frag_a(const __bf16* rowptr, int half) {
    bf16x8 lo = *(const bf16x8*)(rowptr + half * 8);
    bf16x8 hi = *(const bf16x8*)(rowptr + half * 8 + 16);
    return __builtin_shufflevector(lo, hi, 0,1,2,3,4,5,6,7,8,9,10,11,12,13,14,15);
}

// Same A layout, gathered from fp32 (global or LDS) with on-the-fly bf16 convert.
static __device__ __forceinline__ bf16x16 load_frag_a_f32(const float* rowptr, int half) {
    bf16x16 a;
#pragma unroll
    for (int i = 0; i < 8; ++i) {
        a[i]     = (__bf16)rowptr[half * 8 + i];
        a[i + 8] = (__bf16)rowptr[half * 8 + 16 + i];
    }
    return a;
}

// B-fragment (32x16): lane holds col N=lane%16; K runs {h*8..h*8+7, h*8+16..h*8+23}, h=lane/16.
static __device__ __forceinline__ bf16x16 load_frag_b(const __bf16* colptr, int half) {
    bf16x8 lo = *(const bf16x8*)(colptr + half * 8);
    bf16x8 hi = *(const bf16x8*)(colptr + half * 8 + 16);
    return __builtin_shufflevector(lo, hi, 0,1,2,3,4,5,6,7,8,9,10,11,12,13,14,15);
}

// ---------------------------------------------------------------------------
// CDNA5 async global->LDS copy (16B per lane, tracked by ASYNCcnt)
// ---------------------------------------------------------------------------
static __device__ __forceinline__ void async_load_b128(uint32_t lds_byte_off,
                                                       const void* gaddr) {
    // GV mode: vdst = LDS byte offset, vaddr = 64-bit global address
    asm volatile("global_load_async_to_lds_b128 %0, %1, off"
                 :: "v"(lds_byte_off), "v"(gaddr) : "memory");
}
static __device__ __forceinline__ void wait_asynccnt0() {
    asm volatile("s_wait_asynccnt 0x0" ::: "memory");
}
// Generic LDS pointer -> byte offset within the wave's LDS allocation
// (ISA: flat LDS aperture uses addr[31:0] as the in-allocation offset).
static __device__ __forceinline__ uint32_t lds_offset_of(const void* p) {
    return (uint32_t)(uintptr_t)p;
}

// ---------------------------------------------------------------------------
// Problem constants (fixed by setup_inputs)
// ---------------------------------------------------------------------------
#define C_TOT 192
#define HEADS 6
#define HC    32      // channels per head (== WMMA K!)
#define WSZ   7
#define HW    49      // positions per window
#define IMG   224
#define NPIX  (IMG * IMG)

// Dynamic LDS layout (region-aliased, 62208 B total):
//   [0,      24576) : xwin  [64][192] bf16   (phases 0-1)
//                     later: qT[64][32], kT[64][32], vS[32][64], attnT[64][64] (phases 3-5)
//   [24576,  43392) : tmp   [3][49][32] f32  (qkv after GEMM+bias)
//   [43392,  62208) : tmp2  [3][49][32] f32  (after depthwise conv)
#define SMEM_BYTES (24576 + 18816 + 18816)

__global__ __launch_bounds__(128) void win_attn_kernel(
    const float* __restrict__ x,     const float* __restrict__ w_qkv,
    const float* __restrict__ b_qkv, const float* __restrict__ w_dw,
    const float* __restrict__ b_dw,  const float* __restrict__ temperature,
    float* __restrict__ out)
{
    extern __shared__ char smem[];
    __bf16* xwin  = (__bf16*)smem;                    // [64][192]
    float*  tmp   = (float*) (smem + 24576);          // [3][49][32]
    float*  tmp2  = (float*) (smem + 24576 + 18816);  // [3][49][32]
    __bf16* qT    = (__bf16*)smem;                    // [64][32]  pos-major, ch contiguous
    __bf16* kT    = (__bf16*)(smem + 4096);           // [64][32]
    __bf16* vS    = (__bf16*)(smem + 8192);           // [32][64]  ch-major, pos contiguous
    __bf16* attnT = (__bf16*)(smem + 12288);          // [64][64]  m-major, n contiguous

    const int tid  = threadIdx.x;
    const int lane = tid & 31;
    const int wv   = tid >> 5;          // 4 waves
    const int half = lane >> 4;
    const int hi8  = (lane & 16) ? 8 : 0;
    const int head = blockIdx.y;
    const int b    = blockIdx.z;
    const int wy   = blockIdx.x >> 5, wx = blockIdx.x & 31;
    const int y0   = wy * WSZ, x0 = wx * WSZ;
    const float* xb = x + (size_t)b * C_TOT * NPIX;

    // ---- phase 0: stage x window as bf16, B-fragment layout [pos][c] -------
    for (int e = tid; e < 64 * C_TOT; e += 128) {
        int c = e >> 6, pos = e & 63;
        float v = 0.0f;
        if (pos < HW)
            v = xb[(size_t)c * NPIX + (y0 + pos / 7) * IMG + (x0 + pos % 7)];
        xwin[pos * C_TOT + c] = (__bf16)v;
    }
    __syncthreads();

    // ---- phase 1: qkv GEMM for this head: (96 x 49) = W_qkv_rows x x_window
    // wave wv owns position tile [wv*16, wv*16+16); 6 M-tiles (3*32 channels).
    {
        f32x8 acc[6];
#pragma unroll
        for (int m = 0; m < 6; ++m) acc[m] = f32x8_zero();
        const __bf16* bcol = &xwin[(wv * 16 + (lane & 15)) * C_TOT];
        for (int k0 = 0; k0 < C_TOT; k0 += 32) {
            bf16x16 bfrag = load_frag_b(bcol + k0, half);
#pragma unroll
            for (int m = 0; m < 6; ++m) {
                int s = m >> 1;
                int o = s * C_TOT + head * HC + (m & 1) * 16 + (lane & 15);
                bf16x16 afrag = load_frag_a_f32(&w_qkv[(size_t)o * C_TOT + k0], half);
                acc[m] = wmma_bf16(afrag, bfrag, acc[m]);
            }
        }
        int pos = wv * 16 + (lane & 15);
        if (pos < HW) {
#pragma unroll
            for (int m = 0; m < 6; ++m) {
#pragma unroll
                for (int r = 0; r < 8; ++r) {
                    int row = m * 16 + r + hi8;      // 0..95
                    int s = row >> 5, c = row & 31;
                    tmp[(s * HW + pos) * HC + c] =
                        acc[m][r] + b_qkv[s * C_TOT + head * HC + c];
                }
            }
        }
    }
    __syncthreads();

    // ---- phase 2: depthwise 3x3 inside the window (zero pad at edges) ------
    for (int e = tid; e < 3 * HC * HW; e += 128) {
        int s   = e / (HC * HW);
        int rem = e - s * HC * HW;
        int c   = rem / HW;
        int pos = rem - c * HW;
        int my = pos / 7, mx = pos % 7;
        int og = s * C_TOT + head * HC + c;
        const float* wd = &w_dw[og * 9];
        float sum = b_dw[og];
#pragma unroll
        for (int ky = -1; ky <= 1; ++ky)
#pragma unroll
            for (int kx = -1; kx <= 1; ++kx) {
                int yy = my + ky, xx = mx + kx;
                if (yy >= 0 && yy < 7 && xx >= 0 && xx < 7)
                    sum += wd[(ky + 1) * 3 + (kx + 1)] *
                           tmp[(s * HW + yy * 7 + xx) * HC + c];
            }
        tmp2[(s * HW + pos) * HC + c] = sum;
    }
    __syncthreads();

    // ---- phase 3: L2-normalize q,k per position; pack q,k,v as bf16 --------
    {
        int sqk = tid >> 6;                 // 0 -> q, 1 -> k
        int pos = tid & 63;
        __bf16* dst = sqk ? kT : qT;
        if (pos < HW) {
            const float* src = &tmp2[(sqk * HW + pos) * HC];
            float ss = 0.0f;
#pragma unroll
            for (int c = 0; c < HC; ++c) ss += src[c] * src[c];
            float scale = 1.0f / fmaxf(sqrtf(ss), 1e-12f);
#pragma unroll
            for (int c = 0; c < HC; ++c) dst[pos * HC + c] = (__bf16)(src[c] * scale);
        } else {
#pragma unroll
            for (int c = 0; c < HC; ++c) dst[pos * HC + c] = (__bf16)0.0f;
        }
        for (int e = tid; e < HC * 64; e += 128) {
            int c = e >> 6, n = e & 63;
            vS[c * 64 + n] = (n < HW) ? (__bf16)tmp2[(2 * HW + n) * HC + c]
                                      : (__bf16)0.0f;
        }
    }
    __syncthreads();

    // ---- phase 4: attn = k^T q (49x49, K=32 -> single WMMA per tile),
    //               *temperature, softmax over n (axis=-2), store bf16 -------
    const float temp = temperature[head];
    {
        int m = wv * 16 + (lane & 15);      // this wave owns one m-tile
        bf16x16 qfrag = load_frag_b(&qT[m * HC], half);
        f32x8 aacc[4];
#pragma unroll
        for (int tn = 0; tn < 4; ++tn) {
            aacc[tn] = f32x8_zero();
            bf16x16 kfrag = load_frag_a(&kT[(tn * 16 + (lane & 15)) * HC], half);
            aacc[tn] = wmma_bf16(kfrag, qfrag, aacc[tn]);
        }
        float mx = -3.0e38f;
#pragma unroll
        for (int tn = 0; tn < 4; ++tn)
#pragma unroll
            for (int r = 0; r < 8; ++r) {
                int n = tn * 16 + r + hi8;
                float v = aacc[tn][r] * temp;
                if (n >= HW) v = -3.0e38f;   // mask padded k-rows
                aacc[tn][r] = v;
                mx = fmaxf(mx, v);
            }
        mx = fmaxf(mx, __shfl_xor(mx, 16, 32));   // other lane-half of this column
        float sum = 0.0f;
#pragma unroll
        for (int tn = 0; tn < 4; ++tn)
#pragma unroll
            for (int r = 0; r < 8; ++r) {
                float v = __expf(aacc[tn][r] - mx);
                aacc[tn][r] = v;
                sum += v;
            }
        sum += __shfl_xor(sum, 16, 32);
        float inv = 1.0f / sum;
#pragma unroll
        for (int tn = 0; tn < 4; ++tn)
#pragma unroll
            for (int r = 0; r < 8; ++r) {
                int n = tn * 16 + r + hi8;
                attnT[m * 64 + n] = (__bf16)(aacc[tn][r] * inv);
            }
    }
    __syncthreads();

    // ---- phase 5: out = v(32x49) @ attn(49x49); scatter to planar layout ---
    {
        int m = wv * 16 + (lane & 15);
        f32x8 oacc[2] = { f32x8_zero(), f32x8_zero() };
#pragma unroll
        for (int kk = 0; kk < 2; ++kk) {
            bf16x16 bfrag = load_frag_b(&attnT[m * 64 + kk * 32], half);
#pragma unroll
            for (int tc = 0; tc < 2; ++tc) {
                bf16x16 vfrag =
                    load_frag_a(&vS[(tc * 16 + (lane & 15)) * 64 + kk * 32], half);
                oacc[tc] = wmma_bf16(vfrag, bfrag, oacc[tc]);
            }
        }
        if (m < HW) {
            int my = m / 7, mxp = m % 7;
            float* ob = out + (size_t)b * C_TOT * NPIX;
#pragma unroll
            for (int tc = 0; tc < 2; ++tc)
#pragma unroll
                for (int r = 0; r < 8; ++r) {
                    int cg = head * HC + tc * 16 + r + hi8;
                    ob[(size_t)cg * NPIX + (y0 + my) * IMG + (x0 + mxp)] = oacc[tc][r];
                }
        }
    }
}

// ---------------------------------------------------------------------------
// Projection GEMM: Y[b,o,p] = sum_c Wp[o,c] * X[b,c,p] + bias[o]
// Block tile 64(M) x 128(N), K staged 32 at a time; 8 waves, each wave a
// 16x64 strip (4 WMMAs per K-step, A-frag reused).
//   A tile: fp32, copied global->LDS with CDNA5 async DMA (ASYNCcnt).
//   B tile: fp32->bf16 converted through VGPRs (needs a c<->p transpose).
//   Next B tile warmed into L2 with global_prefetch_b8.
// ---------------------------------------------------------------------------
#define GEMM_BM 64
#define GEMM_BN 128

__global__ __launch_bounds__(256) void proj_gemm_kernel(
    const float* __restrict__ X, const float* __restrict__ W,
    const float* __restrict__ bias, float* __restrict__ Y,
    int M, int K, int Np)
{
    __shared__ float  Asf[GEMM_BM][32];  // [m][k] fp32 (async-copied)
    __shared__ __bf16 Bs[GEMM_BN][32];   // [p][k] bf16 (B-fragment friendly)

    const int tid  = threadIdx.x;
    const int lane = tid & 31;
    const int wv   = tid >> 5;
    const int half = lane >> 4;
    const int mi = wv & 3, nj = wv >> 2;
    const int m0 = blockIdx.y * GEMM_BM;
    const int p0 = blockIdx.x * GEMM_BN;
    const float* Xb = X + (size_t)blockIdx.z * K * Np;
    float*       Yb = Y + (size_t)blockIdx.z * M * Np;

    const uint32_t as_base = lds_offset_of(&Asf[0][0]);

    f32x8 acc[4];
#pragma unroll
    for (int j = 0; j < 4; ++j) acc[j] = f32x8_zero();

    for (int k0 = 0; k0 < K; k0 += 32) {
        __syncthreads();   // previous iteration's fragments fully consumed

        // A tile: 64x32 fp32 = 512 x 16B chunks; 2 async copies per thread.
        // Chunk q covers floats [q*4, q*4+4) of the row-major [m][k] tile,
        // which are K-contiguous in W as well -> direct 16B DMA.
#pragma unroll
        for (int q = tid; q < (GEMM_BM * 32) / 4; q += 256) {
            int lin = q * 4;
            int r = lin >> 5, c = lin & 31;
            async_load_b128(as_base + (uint32_t)lin * 4,
                            &W[(size_t)(m0 + r) * K + k0 + c]);
        }

        // B tile: transpose c<->p at 4B granularity, convert fp32->bf16.
        for (int e = tid; e < GEMM_BN * 32; e += 256) {
            int c = e >> 7, p = e & 127;      // consecutive threads -> consecutive p
            Bs[p][c] = (__bf16)Xb[(size_t)(k0 + c) * Np + p0 + p];
        }

        // Warm L2 for the next B tile (1 cacheline-granular probe per thread).
        if (k0 + 32 < K && tid < 128) {
            int c = tid >> 2, seg = tid & 3;
            __builtin_prefetch(&Xb[(size_t)(k0 + 32 + c) * Np + p0 + seg * 32], 0, 0);
        }

        wait_asynccnt0();       // async DMA completion (ASYNCcnt)
        __syncthreads();        // + ds stores visible block-wide

        bf16x16 afrag = load_frag_a_f32(&Asf[mi * 16 + (lane & 15)][0], half);
#pragma unroll
        for (int j = 0; j < 4; ++j) {
            bf16x16 bfrag = load_frag_b(&Bs[nj * 64 + j * 16 + (lane & 15)][0], half);
            acc[j] = wmma_bf16(afrag, bfrag, acc[j]);
        }
    }

    const int mbase = m0 + mi * 16 + ((lane & 16) ? 8 : 0);
    float bv[8];
#pragma unroll
    for (int r = 0; r < 8; ++r) bv[r] = bias[mbase + r];
#pragma unroll
    for (int j = 0; j < 4; ++j)
#pragma unroll
        for (int r = 0; r < 8; ++r)
            Yb[(size_t)(mbase + r) * Np + p0 + nj * 64 + j * 16 + (lane & 15)] =
                acc[j][r] + bv[r];
}

// ---------------------------------------------------------------------------
extern "C" void kernel_launch(void* const* d_in, const int* in_sizes, int n_in,
                              void* d_out, int out_size, void* d_ws, size_t ws_size,
                              hipStream_t stream)
{
    const float* x           = (const float*)d_in[0];
    const float* w_qkv       = (const float*)d_in[1];
    const float* b_qkv       = (const float*)d_in[2];
    const float* w_dw        = (const float*)d_in[3];
    const float* b_dw        = (const float*)d_in[4];
    const float* w_proj      = (const float*)d_in[5];
    const float* b_proj      = (const float*)d_in[6];
    const float* temperature = (const float*)d_in[7];

    float* attnout = (float*)d_ws;      // (4,192,224,224) f32 scratch
    float* outp    = (float*)d_out;

    // Fused qkv-GEMM + window depthwise + normalized attention, per (window, head)
    dim3 g1(32 * 32, HEADS, 4), t1(128);
    win_attn_kernel<<<g1, t1, SMEM_BYTES, stream>>>(
        x, w_qkv, b_qkv, w_dw, b_dw, temperature, attnout);

    // Projection GEMM over planar pixels: M=192, K=192, N=50176 per image
    dim3 g2(NPIX / GEMM_BN, C_TOT / GEMM_BM, 4), t2(256);
    proj_gemm_kernel<<<g2, t2, 0, stream>>>(attnout, w_proj, b_proj, outp,
                                            C_TOT, C_TOT, NPIX);
}
